// SelfAttention_1443109012231
// MI455X (gfx1250) — matrix operand order repaired and checked by reading, not compiled, not run on hardware
//
#include <hip/hip_runtime.h>
#include <hip/hip_bf16.h>
#include <math.h>

// ---------------------------------------------------------------------------
// BERT self-attention block for MI455X (gfx1250, wave32, WMMA 16x16x32 f16)
//   B=4, S=2048, HID=768, NH=12, DH=64
// Async global->LDS double-buffered staging feeding v_wmma.
// ---------------------------------------------------------------------------

typedef _Float16 f16;
typedef __attribute__((ext_vector_type(16))) _Float16 v16h;
typedef __attribute__((ext_vector_type(8)))  _Float16 v8h;
typedef __attribute__((ext_vector_type(8)))  float    v8f;
typedef __attribute__((ext_vector_type(4)))  int      v4i;

#define BB   4
#define SS   2048
#define HIDD 768
#define NHH  12
#define DHH  64
#define MTOT (BB*SS)          // 8192 rows

// ---- CDNA5 async global->LDS support --------------------------------------
#if defined(__has_builtin)
#if __has_builtin(__builtin_amdgcn_global_load_async_to_lds_b128)
#define HAVE_ASYNC_LDS 1
#endif
#endif
#ifndef HAVE_ASYNC_LDS
#define HAVE_ASYNC_LDS 0
#endif

typedef __attribute__((address_space(1))) v4i gv4i;   // global int4
typedef __attribute__((address_space(3))) v4i lv4i;   // LDS int4

// 16B global -> LDS copy (async on CDNA5; synchronous fallback otherwise)
__device__ __forceinline__ void copy_b128(const f16* g, f16* l) {
#if HAVE_ASYNC_LDS
  __builtin_amdgcn_global_load_async_to_lds_b128((gv4i*)g, (lv4i*)l, 0, 0);
#else
  *(v8h*)l = *(const v8h*)g;
#endif
}

__device__ __forceinline__ void wait_async_done() {
#if HAVE_ASYNC_LDS
#if __has_builtin(__builtin_amdgcn_s_wait_asynccnt)
  __builtin_amdgcn_s_wait_asynccnt(0);
#else
  asm volatile("s_wait_asynccnt 0x0" ::: "memory");
#endif
#endif
}

// ---- WMMA helpers ---------------------------------------------------------
// D = A(16x32 f16) * B(32x16 f16) + C(16x16 f32)
__device__ __forceinline__ v8f wmma_f16(v16h a, v16h b, v8f c) {
  return __builtin_amdgcn_wmma_f32_16x16x32_f16(false, a, false, b,
                                                (short)0, c, false, false);
}

__device__ __forceinline__ v16h make_v16(const f16* p0, const f16* p1) {
  v8h lo = *(const v8h*)p0;
  v8h hi = *(const v8h*)p1;
  v16h r;
#pragma unroll
  for (int i = 0; i < 8; ++i) { r[i] = lo[i]; r[8 + i] = hi[i]; }
  return r;
}
__device__ __forceinline__ v16h load_v16(const f16* p) { return make_v16(p, p + 8); }

__device__ __forceinline__ v8f zero_v8f() {
  v8f z;
#pragma unroll
  for (int i = 0; i < 8; ++i) z[i] = 0.0f;
  return z;
}

// ---------------------------------------------------------------------------
// Prep kernels
// ---------------------------------------------------------------------------
__global__ void cvt_f32_f16(const float* __restrict__ in, f16* __restrict__ out, int n) {
  int i = blockIdx.x * blockDim.x + threadIdx.x;
  if (i < n) out[i] = (f16)in[i];
}

// Wt[n*768 + k] = (f16) W[k*768 + n]   (768x768)
__global__ void transpose_cvt(const float* __restrict__ W, f16* __restrict__ Wt) {
  int i = blockIdx.x * blockDim.x + threadIdx.x;
  int n = i / HIDD, k = i - n * HIDD;
  Wt[n * HIDD + k] = (f16)W[k * HIDD + n];
}

// ---------------------------------------------------------------------------
// QKV projection GEMM:  out = scale * (xh @ W + bias)
//   vtrans=0: out(B,NH,S,DH)   vtrans=1: out(B,NH,DH,S)  (for V)
//   grid = (MTOT/128, HIDD/64), block = 256 (8 waves, 16 M-rows each)
// ---------------------------------------------------------------------------
__global__ __launch_bounds__(256) void qkv_gemm(
    const f16* __restrict__ xh, const f16* __restrict__ Wt,
    const float* __restrict__ bias, f16* __restrict__ out,
    float scale, int vtrans) {
  __shared__ __align__(16) f16 lds_w[2][64 * 32];

  const int tid  = threadIdx.x;
  const int wave = tid >> 5;
  const int lane = tid & 31;
  const int lh   = lane & 15;
  const int hs   = lane >> 4;
  const int mbase = blockIdx.x * 128 + wave * 16;
  const int nbase = blockIdx.y * 64;
  const f16* xrow = xh + (size_t)(mbase + lh) * HIDD;

  // per-thread staging slice: 64 cols x 32 k, one b128 each
  const int cn  = tid >> 2;
  const int cko = (tid & 3) * 8;
  const f16* wsrc = &Wt[(size_t)(nbase + cn) * HIDD + cko];
  f16* wdst0 = &lds_w[0][cn * 32 + cko];
  f16* wdst1 = &lds_w[1][cn * 32 + cko];

  v8f acc[4];
#pragma unroll
  for (int t = 0; t < 4; ++t) acc[t] = zero_v8f();

  const int NSTEP = HIDD / 32;   // 24
  copy_b128(wsrc, wdst0);        // preload tile 0
#pragma unroll 1
  for (int step = 0; step < NSTEP; ++step) {
    const int kb  = step * 32;
    const int cur = step & 1;
    wait_async_done();
    __syncthreads();             // buf[cur] filled & previous compute finished
    if (step + 1 < NSTEP)
      copy_b128(wsrc + kb + 32, cur ? wdst0 : wdst1);

    v16h a = make_v16(xrow + kb + hs * 16, xrow + kb + hs * 16 + 8);
#pragma unroll
    for (int t = 0; t < 4; ++t) {
      v16h b = load_v16(&lds_w[cur][(t * 16 + lh) * 32 + hs * 16]);
      acc[t] = wmma_f16(a, b, acc[t]);
    }
  }

#pragma unroll
  for (int t = 0; t < 4; ++t) {
    int col = nbase + t * 16 + lh;
    float bv = bias[col];
    int h = col >> 6, d = col & 63;
#pragma unroll
    for (int r = 0; r < 8; ++r) {
      int m = mbase + r + 8 * hs;
      int b = m >> 11, s = m & (SS - 1);
      f16 hv = (f16)((acc[t][r] + bv) * scale);
      if (vtrans)
        out[(((size_t)(b * NHH + h)) * DHH + d) * SS + s] = hv;
      else
        out[(((size_t)(b * NHH + h)) * SS + s) * DHH + d] = hv;
    }
  }
}

// ---------------------------------------------------------------------------
// Flash attention:  ctx(B,S,HID) f16
//   Q,K: (B,NH,S,DH)  Vt: (B,NH,DH,S)
//   grid = (S/128, B*NH), block = 256; each wave owns 16 query rows.
// ---------------------------------------------------------------------------
__global__ __launch_bounds__(256) void attn_kernel(
    const f16* __restrict__ Q, const f16* __restrict__ K, const f16* __restrict__ Vt,
    const float* __restrict__ mask, f16* __restrict__ ctxh) {
  __shared__ __align__(16) f16 lds_k [2][32 * 64];    // [key][d]
  __shared__ __align__(16) f16 lds_vt[2][64 * 32];    // [d][key]
  __shared__ __align__(16) f16 lds_p [8 * 16 * 32];   // per-wave [qrow][key]

  const int tid  = threadIdx.x;
  const int wave = tid >> 5;
  const int lane = tid & 31;
  const int lh   = lane & 15;
  const int hs   = lane >> 4;
  const int bh   = blockIdx.y;
  const int b    = bh / NHH;
  const int h    = bh - b * NHH;
  const int q0   = blockIdx.x * 128 + wave * 16;

  const f16* Qb  = Q  + (size_t)bh * SS * DHH;
  const f16* Kb  = K  + (size_t)bh * SS * DHH;
  const f16* Vtb = Vt + (size_t)bh * DHH * SS;

  // per-thread staging slices (one b128 each for K and Vt)
  const int kk = tid >> 3, kdo = (tid & 7) * 8;       // K: 32 keys x 64 d
  const int vd = tid >> 2, vko = (tid & 3) * 8;       // Vt: 64 d x 32 keys

  // Q fragments (d = 0..31 and 32..63), Q pre-scaled by 1/sqrt(DH)
  const f16* qrow = Qb + (size_t)(q0 + lh) * DHH;
  v16h qa0 = make_v16(qrow + hs * 16,      qrow + hs * 16 + 8);
  v16h qa1 = make_v16(qrow + 32 + hs * 16, qrow + 32 + hs * 16 + 8);

  float m8[8], l8[8];
  v8f ctx[4];
#pragma unroll
  for (int r = 0; r < 8; ++r) { m8[r] = -1.0e30f; l8[r] = 0.0f; }
#pragma unroll
  for (int t = 0; t < 4; ++t) ctx[t] = zero_v8f();

  f16* pbuf = &lds_p[wave * 16 * 32];

  // preload key-chunk 0
  copy_b128(&Kb[(size_t)kk * DHH + kdo],  &lds_k[0][kk * 64 + kdo]);
  copy_b128(&Vtb[(size_t)vd * SS + vko],  &lds_vt[0][vd * 32 + vko]);

  const int NSTEP = SS / 32;   // 64
#pragma unroll 1
  for (int step = 0; step < NSTEP; ++step) {
    const int kb  = step * 32;
    const int cur = step & 1;
    wait_async_done();
    __syncthreads();
    if (step + 1 < NSTEP) {
      int nkb = kb + 32;
      copy_b128(&Kb[(size_t)(nkb + kk) * DHH + kdo], &lds_k[cur ^ 1][kk * 64 + kdo]);
      copy_b128(&Vtb[(size_t)vd * SS + nkb + vko],   &lds_vt[cur ^ 1][vd * 32 + vko]);
    }

    // scores: S[16q x 32k] = Q(16x64) @ K^T, two 16-col tiles
    v8f st[2];
#pragma unroll
    for (int t = 0; t < 2; ++t) {
      v8f z = zero_v8f();
      v16h b0 = load_v16(&lds_k[cur][(t * 16 + lh) * 64 + hs * 16]);
      z = wmma_f16(qa0, b0, z);
      v16h b1 = load_v16(&lds_k[cur][(t * 16 + lh) * 64 + 32 + hs * 16]);
      z = wmma_f16(qa1, b1, z);
      float mk = mask[b * SS + kb + t * 16 + lh];
      float biasv = (1.0f - mk) * -10000.0f;
#pragma unroll
      for (int r = 0; r < 8; ++r) z[r] += biasv;
      st[t] = z;
    }

    // row max over 32 keys (reduce across the 16-lane column group)
    float rmax[8];
#pragma unroll
    for (int r = 0; r < 8; ++r) rmax[r] = fmaxf(st[0][r], st[1][r]);
#pragma unroll
    for (int off = 1; off < 16; off <<= 1)
#pragma unroll
      for (int r = 0; r < 8; ++r)
        rmax[r] = fmaxf(rmax[r], __shfl_xor(rmax[r], off, 32));

    float alpha[8];
#pragma unroll
    for (int r = 0; r < 8; ++r) {
      float mn = fmaxf(m8[r], rmax[r]);
      alpha[r] = __expf(m8[r] - mn);
      m8[r] = mn;
    }

    float rsum[8];
#pragma unroll
    for (int r = 0; r < 8; ++r) rsum[r] = 0.0f;
#pragma unroll
    for (int t = 0; t < 2; ++t)
#pragma unroll
      for (int r = 0; r < 8; ++r) {
        float p = __expf(st[t][r] - m8[r]);
        st[t][r] = p;
        rsum[r] += p;
      }
#pragma unroll
    for (int off = 1; off < 16; off <<= 1)
#pragma unroll
      for (int r = 0; r < 8; ++r) rsum[r] += __shfl_xor(rsum[r], off, 32);
#pragma unroll
    for (int r = 0; r < 8; ++r) l8[r] = l8[r] * alpha[r] + rsum[r];

    // rescale running context
#pragma unroll
    for (int t = 0; t < 4; ++t)
#pragma unroll
      for (int r = 0; r < 8; ++r) ctx[t][r] *= alpha[r];

    // P: C-layout -> LDS -> A-layout (per-wave private region)
#pragma unroll
    for (int t = 0; t < 2; ++t)
#pragma unroll
      for (int r = 0; r < 8; ++r)
        pbuf[(r + 8 * hs) * 32 + t * 16 + lh] = (f16)st[t][r];
    v16h pa = make_v16(&pbuf[lh * 32 + hs * 16], &pbuf[lh * 32 + hs * 16 + 8]);

    // ctx += P(16x32) @ V(32x64)
#pragma unroll
    for (int t = 0; t < 4; ++t) {
      v16h bv = load_v16(&lds_vt[cur][(t * 16 + lh) * 32 + hs * 16]);
      ctx[t] = wmma_f16(pa, bv, ctx[t]);
    }
  }

  // finalize: ctx / l, store (B,S,HID) f16
  float inv[8];
#pragma unroll
  for (int r = 0; r < 8; ++r) inv[r] = (l8[r] > 0.0f) ? (1.0f / l8[r]) : 0.0f;
#pragma unroll
  for (int t = 0; t < 4; ++t)
#pragma unroll
    for (int r = 0; r < 8; ++r) {
      int s = q0 + r + 8 * hs;
      ctxh[((size_t)(b * SS + s)) * HIDD + h * DHH + t * 16 + lh] =
          (f16)(ctx[t][r] * inv[r]);
    }
}

// ---------------------------------------------------------------------------
// Output projection + bias + residual:  y(f32) = ctx @ Wo + bo + x
//   grid = (MTOT/128, HIDD/64)
// ---------------------------------------------------------------------------
__global__ __launch_bounds__(256) void out_proj(
    const f16* __restrict__ ctxh, const f16* __restrict__ Wot,
    const float* __restrict__ bo, const float* __restrict__ xres,
    float* __restrict__ y) {
  __shared__ __align__(16) f16 lds_w[2][64 * 32];

  const int tid  = threadIdx.x;
  const int wave = tid >> 5;
  const int lane = tid & 31;
  const int lh   = lane & 15;
  const int hs   = lane >> 4;
  const int mbase = blockIdx.x * 128 + wave * 16;
  const int nbase = blockIdx.y * 64;
  const f16* arow = ctxh + (size_t)(mbase + lh) * HIDD;

  const int cn  = tid >> 2;
  const int cko = (tid & 3) * 8;
  const f16* wsrc = &Wot[(size_t)(nbase + cn) * HIDD + cko];
  f16* wdst0 = &lds_w[0][cn * 32 + cko];
  f16* wdst1 = &lds_w[1][cn * 32 + cko];

  v8f acc[4];
#pragma unroll
  for (int t = 0; t < 4; ++t) acc[t] = zero_v8f();

  const int NSTEP = HIDD / 32;
  copy_b128(wsrc, wdst0);
#pragma unroll 1
  for (int step = 0; step < NSTEP; ++step) {
    const int kb  = step * 32;
    const int cur = step & 1;
    wait_async_done();
    __syncthreads();
    if (step + 1 < NSTEP)
      copy_b128(wsrc + kb + 32, cur ? wdst0 : wdst1);

    v16h a = make_v16(arow + kb + hs * 16, arow + kb + hs * 16 + 8);
#pragma unroll
    for (int t = 0; t < 4; ++t) {
      v16h b = load_v16(&lds_w[cur][(t * 16 + lh) * 32 + hs * 16]);
      acc[t] = wmma_f16(a, b, acc[t]);
    }
  }

#pragma unroll
  for (int t = 0; t < 4; ++t) {
    int col = nbase + t * 16 + lh;
    float bv = bo[col];
#pragma unroll
    for (int r = 0; r < 8; ++r) {
      size_t m = mbase + r + 8 * hs;
      y[m * HIDD + col] = acc[t][r] + bv + xres[m * HIDD + col];
    }
  }
}

// ---------------------------------------------------------------------------
// In-place LayerNorm over rows of y (768 cols); wave per row, 8 rows/block.
// ---------------------------------------------------------------------------
__global__ __launch_bounds__(256) void layernorm(
    float* __restrict__ y, const float* __restrict__ g, const float* __restrict__ bta) {
  const int wave = threadIdx.x >> 5;
  const int lane = threadIdx.x & 31;
  const int row  = blockIdx.x * 8 + wave;
  float* p = y + (size_t)row * HIDD;

  float vals[24];
  float sum = 0.0f;
#pragma unroll
  for (int i = 0; i < 24; ++i) { vals[i] = p[lane + i * 32]; sum += vals[i]; }
#pragma unroll
  for (int off = 1; off < 32; off <<= 1) sum += __shfl_xor(sum, off, 32);
  float mu = sum * (1.0f / 768.0f);

  float vs = 0.0f;
#pragma unroll
  for (int i = 0; i < 24; ++i) { float d = vals[i] - mu; vs += d * d; }
#pragma unroll
  for (int off = 1; off < 32; off <<= 1) vs += __shfl_xor(vs, off, 32);
  float rstd = rsqrtf(vs * (1.0f / 768.0f) + 1e-5f);

#pragma unroll
  for (int i = 0; i < 24; ++i) {
    int c = lane + i * 32;
    p[c] = (vals[i] - mu) * rstd * g[c] + bta[c];
  }
}

// ---------------------------------------------------------------------------
// Host launcher
// ---------------------------------------------------------------------------
extern "C" void kernel_launch(void* const* d_in, const int* in_sizes, int n_in,
                              void* d_out, int out_size, void* d_ws, size_t ws_size,
                              hipStream_t stream) {
  const float* x    = (const float*)d_in[0];
  const float* mask = (const float*)d_in[1];
  const float* Wq   = (const float*)d_in[2];
  const float* bq   = (const float*)d_in[3];
  const float* Wk   = (const float*)d_in[4];
  const float* bk   = (const float*)d_in[5];
  const float* Wv   = (const float*)d_in[6];
  const float* bv   = (const float*)d_in[7];
  const float* Wo   = (const float*)d_in[8];
  const float* bo   = (const float*)d_in[9];
  const float* ln_g = (const float*)d_in[10];
  const float* ln_b = (const float*)d_in[11];
  float* out = (float*)d_out;

  char* ws = (char*)d_ws;
  size_t off = 0;
  auto carve = [&](size_t bytes) -> void* {
    void* p = ws + off;
    off += (bytes + 255) & ~(size_t)255;
    return p;
  };
  const size_t XSZ = (size_t)MTOT * HIDD;           // 6,291,456 elements
  const size_t WSZ = (size_t)HIDD * HIDD;           //   589,824 elements
  f16* xh   = (f16*)carve(XSZ * 2);
  f16* Wqt  = (f16*)carve(WSZ * 2);
  f16* Wkt  = (f16*)carve(WSZ * 2);
  f16* Wvt  = (f16*)carve(WSZ * 2);
  f16* Wot  = (f16*)carve(WSZ * 2);
  f16* Qd   = (f16*)carve(XSZ * 2);
  f16* Kd   = (f16*)carve(XSZ * 2);
  f16* Vtd  = (f16*)carve(XSZ * 2);
  f16* ctxh = (f16*)carve(XSZ * 2);
  (void)ws_size; (void)in_sizes; (void)n_in; (void)out_size;

  cvt_f32_f16<<<(int)(XSZ / 256), 256, 0, stream>>>(x, xh, (int)XSZ);
  transpose_cvt<<<(int)(WSZ / 256), 256, 0, stream>>>(Wq, Wqt);
  transpose_cvt<<<(int)(WSZ / 256), 256, 0, stream>>>(Wk, Wkt);
  transpose_cvt<<<(int)(WSZ / 256), 256, 0, stream>>>(Wv, Wvt);
  transpose_cvt<<<(int)(WSZ / 256), 256, 0, stream>>>(Wo, Wot);

  dim3 gq(MTOT / 128, HIDD / 64);
  qkv_gemm<<<gq, 256, 0, stream>>>(xh, Wqt, bq, Qd, 0.125f, 0);  // 1/sqrt(64)
  qkv_gemm<<<gq, 256, 0, stream>>>(xh, Wkt, bk, Kd, 1.0f, 0);
  qkv_gemm<<<gq, 256, 0, stream>>>(xh, Wvt, bv, Vtd, 1.0f, 1);   // V transposed

  dim3 ga(SS / 128, BB * NHH);
  attn_kernel<<<ga, 256, 0, stream>>>(Qd, Kd, Vtd, mask, ctxh);

  dim3 go(MTOT / 128, HIDD / 64);
  out_proj<<<go, 256, 0, stream>>>(ctxh, Wot, bo, x, out);

  layernorm<<<MTOT / 8, 256, 0, stream>>>(out, ln_g, ln_b);
}
